// MultiAttention_66073776882075
// MI455X (gfx1250) — matrix operand order repaired and checked by reading, not compiled, not run on hardware
//
#include <hip/hip_runtime.h>

// ---------------------------------------------------------------------------
// Types for CDNA5 WMMA (wave32): bf16 A/B fragments (16 elems = 8 VGPRs),
// f32 C/D accumulator (8 elems = 8 VGPRs).
// ---------------------------------------------------------------------------
typedef __bf16 bf16;
typedef bf16  v4bf  __attribute__((ext_vector_type(4)));
typedef bf16  v8bf  __attribute__((ext_vector_type(8)));
typedef bf16  v16bf __attribute__((ext_vector_type(16)));
typedef float v8f   __attribute__((ext_vector_type(8)));

typedef unsigned int u32x4 __attribute__((ext_vector_type(4)));
typedef int          i32x4 __attribute__((ext_vector_type(4)));
typedef int          i32x8 __attribute__((ext_vector_type(8)));

__device__ __forceinline__ v16bf cat8(v8bf lo, v8bf hi) {
    union { v16bf v; v8bf h[2]; } u;
    u.h[0] = lo; u.h[1] = hi;
    return u.v;
}

__device__ __forceinline__ v8f zero8() {
    v8f z = {0.f, 0.f, 0.f, 0.f, 0.f, 0.f, 0.f, 0.f};
    return z;
}

__device__ __forceinline__ v8f wmma_bf16(v16bf a, v16bf b, v8f c) {
    // D = A(16x32 bf16) x B(32x16 bf16) + C(16x16 f32)
    return __builtin_amdgcn_wmma_f32_16x16x32_bf16(
        false, a, false, b, (short)0, c, false, false);
}

// Low 32 bits of a generic pointer into LDS == byte offset within the
// workgroup's LDS allocation (aperture ABI: addr[31:0] is the LDS offset).
__device__ __forceinline__ uint32_t lds_off(const void* p) {
    return (uint32_t)(uintptr_t)p;
}

// ---------------------------------------------------------------------------
// Tensor Data Mover: 2-D tile (rows x contiguous elems) global -> LDS with
// optional LDS row padding. data_size = 2 bytes (bf16).
//   D# group0: count=1 | lds_addr | global_addr[56:0] | type=2
//   D# group1: data_size, pad ctrl, tensor dims, tile dims, dim0 stride
// ---------------------------------------------------------------------------
__device__ __forceinline__ void tdm_load_2d(uint32_t lds_addr, const void* gaddr,
                                            uint32_t tile_d0, uint32_t tile_d1,
                                            uint32_t tensor_d0, uint32_t tensor_d1,
                                            uint64_t stride0,     // elements
                                            int pad_int_code, int pad_amt_code,
                                            bool pad_en) {
    uint64_t ga = (uint64_t)(uintptr_t)gaddr;
    u32x4 g0;
    g0[0] = 1u;                                                 // count=1
    g0[1] = lds_addr;                                           // bits 63:32
    g0[2] = (uint32_t)ga;                                       // addr[31:0]
    g0[3] = (uint32_t)((ga >> 32) & 0x01FFFFFFu) | (2u << 30);  // addr[56:32], type=2

    uint32_t d0 = (1u << 16);                                   // data_size=1 -> 2B
    if (pad_en)
        d0 |= (1u << 20) | ((uint32_t)pad_int_code << 22) | ((uint32_t)pad_amt_code << 25);
    i32x8 g1;
    g1[0] = (int)d0;
    g1[1] = (int)((tensor_d0 & 0xFFFFu) << 16);                 // tensor_dim0 lo @bit48
    g1[2] = (int)((tensor_d0 >> 16) | ((tensor_d1 & 0xFFFFu) << 16));
    g1[3] = (int)((tensor_d1 >> 16) | (tile_d0 << 16));         // tile_dim0 @bit112
    g1[4] = (int)(tile_d1 & 0xFFFFu);                           // tile_dim1; tile_dim2=0
    g1[5] = (int)(uint32_t)(stride0 & 0xFFFFFFFFu);             // dim0 stride lo
    g1[6] = (int)(uint32_t)((stride0 >> 32) & 0xFFFFu);         // stride hi; stride1=0
    g1[7] = 0;

    i32x4 z4 = {0, 0, 0, 0};
#if defined(__clang_major__) && (__clang_major__ >= 23)
    i32x8 z8 = {0, 0, 0, 0, 0, 0, 0, 0};
    __builtin_amdgcn_tensor_load_to_lds(g0, g1, z4, z4, z8, 0);
#else
    __builtin_amdgcn_tensor_load_to_lds(g0, g1, z4, z4, 0);
#endif
}

// Async global -> LDS, 16 bytes per lane (tracked by ASYNCcnt).
__device__ __forceinline__ void async_ld_b128(uint32_t dst_lds_off, const bf16* g) {
    asm volatile("global_load_async_to_lds_b128 %0, %1, off"
                 :: "v"(dst_lds_off), "v"(g)
                 : "memory");
}
__device__ __forceinline__ void wait_async0() {
    asm volatile("s_wait_asynccnt 0x0" ::: "memory");
}

// ---------------------------------------------------------------------------
// Model dimensions (fixed by the reference)
// ---------------------------------------------------------------------------
#define EMBED 2048
#define HEADS 16
#define HDIM  128
#define BATCH 2
#define SEQ   2048
#define ROWS  (BATCH * SEQ)   // 4096

// ---------------------------------------------------------------------------
// Kernel 1: f32 -> bf16 conversion (vectorized)
// ---------------------------------------------------------------------------
__global__ __launch_bounds__(256)
void cvt_f32_bf16(const float* __restrict__ src, bf16* __restrict__ dst, long n) {
    long i = ((long)blockIdx.x * blockDim.x + threadIdx.x) * 4;
    if (i + 3 < n) {
        float4 v = *(const float4*)(src + i);
        v4bf o = { (bf16)v.x, (bf16)v.y, (bf16)v.z, (bf16)v.w };
        *(v4bf*)(dst + i) = o;
    }
}

// ---------------------------------------------------------------------------
// Kernel 2/5: C[M,N] = A[M,K] . B[N,K]^T   (bf16 in, f32 accumulate)
// Block = 256 threads (8 waves, 4x2), tile 128(M) x 128(N), K-step 64.
// Tiles are moved by the Tensor Data Mover: wave0 issues the A-tile D#,
// wave1 the B-tile D#, double buffered; s_wait_tensorcnt retires the oldest.
// LDS rows padded by TDM (32 DWORDs + 4 DWORD pad -> 72-element stride).
// ---------------------------------------------------------------------------
#define GLDT 72   // padded LDS row stride (64 + 8 elems; 144B, 16B aligned)

template <bool F32OUT>
__global__ __launch_bounds__(256)
void gemm_nt(const bf16* __restrict__ A, const bf16* __restrict__ B,
             void* __restrict__ Cout, int M, int N, int K) {
    __shared__ bf16 As[2][128 * GLDT];
    __shared__ bf16 Bs[2][128 * GLDT];

    const int tid   = threadIdx.x;
    const int lane  = tid & 31;
    const int wave  = tid >> 5;
    const int uwave = __builtin_amdgcn_readfirstlane(wave);  // scalar branch
    const int wm    = wave >> 1;       // 0..3 -> 32-row chunk
    const int wn    = wave & 1;        // 0..1 -> 64-col chunk
    const int ln    = lane & 15;
    const int g     = lane >> 4;
    const int m0    = blockIdx.y * 128;
    const int n0    = blockIdx.x * 128;
    const int nK    = K >> 6;

    v8f acc[2][4];
    #pragma unroll
    for (int i = 0; i < 2; ++i)
        #pragma unroll
        for (int j = 0; j < 4; ++j) acc[i][j] = zero8();

    // Issue TDM for K-tile kt into buffer buf (one descriptor per matrix).
    auto issue = [&](int kt, int buf) {
        if (uwave == 0) {
            tdm_load_2d(lds_off(&As[buf][0]), A + (size_t)m0 * K + kt * 64,
                        /*tile*/64, 128, /*tensor*/(uint32_t)K, (uint32_t)M,
                        (uint64_t)K, /*pad: 32 DW*/4, /*4 DW*/3, true);
        } else if (uwave == 1) {
            tdm_load_2d(lds_off(&Bs[buf][0]), B + (size_t)n0 * K + kt * 64,
                        /*tile*/64, 128, /*tensor*/(uint32_t)K, (uint32_t)N,
                        (uint64_t)K, 4, 3, true);
        }
    };

    issue(0, 0);
    for (int kt = 0; kt < nK; ++kt) {
        const int buf = kt & 1;
        if (kt + 1 < nK) {
            issue(kt + 1, buf ^ 1);
            __builtin_amdgcn_s_wait_tensorcnt(1);  // oldest (tile kt) complete
        } else {
            __builtin_amdgcn_s_wait_tensorcnt(0);
        }
        __syncthreads();                           // tile kt visible to all waves

        const bf16* as = &As[buf][0];
        const bf16* bs = &Bs[buf][0];
        #pragma unroll
        for (int kc = 0; kc < 64; kc += 32) {
            v16bf af[2], bfr[4];
            #pragma unroll
            for (int i = 0; i < 2; ++i) {
                int row = wm * 32 + i * 16 + ln;
                af[i] = cat8(*(const v8bf*)&as[row * GLDT + kc + g * 8],
                             *(const v8bf*)&as[row * GLDT + kc + 16 + g * 8]);
            }
            #pragma unroll
            for (int j = 0; j < 4; ++j) {
                int col = wn * 64 + j * 16 + ln;
                bfr[j] = cat8(*(const v8bf*)&bs[col * GLDT + kc + g * 8],
                              *(const v8bf*)&bs[col * GLDT + kc + 16 + g * 8]);
            }
            #pragma unroll
            for (int i = 0; i < 2; ++i)
                #pragma unroll
                for (int j = 0; j < 4; ++j)
                    acc[i][j] = wmma_bf16(af[i], bfr[j], acc[i][j]);
        }
        __syncthreads();                           // reads done -> buf reusable
    }

    // Epilogue: C-fragment row = r + 8*g, col = lane&15
    #pragma unroll
    for (int i = 0; i < 2; ++i)
        #pragma unroll
        for (int j = 0; j < 4; ++j) {
            int col = n0 + wn * 64 + j * 16 + ln;
            #pragma unroll
            for (int r = 0; r < 8; ++r) {
                int row = m0 + wm * 32 + i * 16 + g * 8 + r;
                float v = acc[i][j][r];
                if (F32OUT) ((float*)Cout)[(size_t)row * N + col] = v;
                else        ((bf16*)Cout)[(size_t)row * N + col] = (bf16)v;
            }
        }
}

// ---------------------------------------------------------------------------
// Kernel 3: RoPE on Q and K (in place, bf16 storage, f32 math)
// ---------------------------------------------------------------------------
__global__ __launch_bounds__(256)
void rope_kernel(bf16* __restrict__ Qb, bf16* __restrict__ Kb) {
    long idx = (long)blockIdx.x * blockDim.x + threadIdx.x;  // over B*S*H*64
    int p = (int)(idx & 63);
    long t = idx >> 6;
    int h = (int)(t & (HEADS - 1)); t >>= 4;
    int s = (int)(t & (SEQ - 1));
    int b = (int)(t >> 11);

    float fr = __expf(-(2.0f * (float)p / (float)HDIM) * 9.210340371976184f); // ln(1e4)
    float sn, cs;
    __sincosf((float)s * fr, &sn, &cs);

    size_t base = (((size_t)(b * SEQ + s)) * HEADS + h) * HDIM;
    bf16* q = Qb + base;
    bf16* k = Kb + base;
    float q0 = (float)q[p], q1 = (float)q[p + 64];
    q[p]      = (bf16)(q0 * cs - q1 * sn);
    q[p + 64] = (bf16)(q1 * cs + q0 * sn);
    float k0 = (float)k[p], k1 = (float)k[p + 64];
    k[p]      = (bf16)(k0 * cs - k1 * sn);
    k[p + 64] = (bf16)(k1 * cs + k0 * sn);
}

// ---------------------------------------------------------------------------
// Kernel 4: causal flash attention. Block = 8 waves = 128 query rows,
// each wave owns 16 rows; key tiles of 32 streamed through LDS.
// K-tile moves via async global->LDS (ASYNCcnt); V transposed via VGPRs.
// ---------------------------------------------------------------------------
#define KS_LD 136   // 32x128 K-tile, row stride 136 elems (272B, 16B aligned)
#define VT_LD 40    // 128x32 V^T tile, row stride 40 elems (80B, 16B aligned)
#define PL_LD 40    // per-wave 16x32 P staging

__global__ __launch_bounds__(256)
void flash_attn(const bf16* __restrict__ Q, const bf16* __restrict__ Kk,
                const bf16* __restrict__ V, const float* __restrict__ maskp,
                bf16* __restrict__ ctx) {
    __shared__ bf16 Ks[32 * KS_LD];
    __shared__ bf16 VTs[128 * VT_LD];
    __shared__ bf16 Pls[8 * 16 * PL_LD];

    const int tid  = threadIdx.x;
    const int lane = tid & 31;
    const int wave = tid >> 5;
    const int ln   = lane & 15;
    const int g    = lane >> 4;
    const int b    = blockIdx.y >> 4;
    const int h    = blockIdx.y & 15;
    const int qbase = blockIdx.x * 128;
    const int q0    = qbase + wave * 16;

    // Persistent Q fragments for this wave's 16 rows (4 K-chunks of 32)
    v16bf qf[4];
    {
        int row = q0 + ln;
        const bf16* qp = Q + ((size_t)(b * SEQ + row)) * EMBED + h * HDIM;
        #pragma unroll
        for (int kc = 0; kc < 4; ++kc)
            qf[kc] = cat8(*(const v8bf*)(qp + kc * 32 + g * 8),
                          *(const v8bf*)(qp + kc * 32 + 16 + g * 8));
    }

    float mi[8], li[8];
    v8f   o[8];
    #pragma unroll
    for (int r = 0; r < 8; ++r) { mi[r] = -3.0e38f; li[r] = 0.f; }
    #pragma unroll
    for (int nf = 0; nf < 8; ++nf) o[nf] = zero8();

    const float scale = 0.08838834764831845f;   // 1/sqrt(128)
    const int kend = qbase + 128;               // causal: keys <= max query

    for (int j0 = 0; j0 < kend; j0 += 32) {
        // K-tile: async global->LDS, 2 x b128 per thread (overlaps V work)
        #pragma unroll
        for (int c = tid; c < 512; c += 256) {
            int kr = c >> 4;
            int c8 = (c & 15) * 8;
            const bf16* kp = Kk + (size_t)(b * SEQ + j0 + kr) * EMBED + h * HDIM + c8;
            async_ld_b128(lds_off(&Ks[kr * KS_LD + c8]), kp);
        }
        // V-tile: load + transpose into LDS while K streams in
        #pragma unroll
        for (int c = tid; c < 512; c += 256) {
            int kr = c >> 4;
            int c8 = (c & 15) * 8;
            v8bf vv = *(const v8bf*)(V + (size_t)(b * SEQ + j0 + kr) * EMBED + h * HDIM + c8);
            #pragma unroll
            for (int i = 0; i < 8; ++i) VTs[(c8 + i) * VT_LD + kr] = vv[i];
        }
        wait_async0();
        __syncthreads();

        // S = Q K^T : two 16x16 accumulators cover 32 keys
        v8f s0 = zero8(), s1 = zero8();
        #pragma unroll
        for (int kc = 0; kc < 4; ++kc) {
            v16bf b0 = cat8(*(const v8bf*)&Ks[ln * KS_LD + kc * 32 + g * 8],
                            *(const v8bf*)&Ks[ln * KS_LD + kc * 32 + 16 + g * 8]);
            v16bf b1 = cat8(*(const v8bf*)&Ks[(16 + ln) * KS_LD + kc * 32 + g * 8],
                            *(const v8bf*)&Ks[(16 + ln) * KS_LD + kc * 32 + 16 + g * 8]);
            s0 = wmma_bf16(qf[kc], b0, s0);
            s1 = wmma_bf16(qf[kc], b1, s1);
        }

        // Online softmax (C layout: row = r + 8g, key col = lane&15 [+16])
        int key0 = j0 + ln, key1 = j0 + 16 + ln;
        float pm0 = (1.0f - maskp[b * SEQ + key0]) * -1e9f;
        float pm1 = (1.0f - maskp[b * SEQ + key1]) * -1e9f;
        float al[8];
        #pragma unroll
        for (int r = 0; r < 8; ++r) {
            int qrow = q0 + g * 8 + r;
            float v0 = s0[r] * scale + pm0 + (key0 > qrow ? -3.0e38f : 0.f);
            float v1 = s1[r] * scale + pm1 + (key1 > qrow ? -3.0e38f : 0.f);
            float rm = fmaxf(v0, v1);
            #pragma unroll
            for (int d = 1; d < 16; d <<= 1) rm = fmaxf(rm, __shfl_xor(rm, d, 16));
            float mn = fmaxf(mi[r], rm);
            float a  = __expf(mi[r] - mn);
            float p0 = __expf(v0 - mn);
            float p1 = __expf(v1 - mn);
            float rs = p0 + p1;
            #pragma unroll
            for (int d = 1; d < 16; d <<= 1) rs += __shfl_xor(rs, d, 16);
            li[r] = li[r] * a + rs;
            mi[r] = mn;
            al[r] = a;
            // Stage P (C layout -> LDS) for A-fragment reload
            Pls[(wave * 16 + g * 8 + r) * PL_LD + ln]      = (bf16)p0;
            Pls[(wave * 16 + g * 8 + r) * PL_LD + 16 + ln] = (bf16)p1;
        }

        // Rescale running output, then O += P x V
        #pragma unroll
        for (int nf = 0; nf < 8; ++nf)
            #pragma unroll
            for (int r = 0; r < 8; ++r) o[nf][r] *= al[r];

        v16bf pf = cat8(*(const v8bf*)&Pls[(wave * 16 + ln) * PL_LD + g * 8],
                        *(const v8bf*)&Pls[(wave * 16 + ln) * PL_LD + 16 + g * 8]);
        #pragma unroll
        for (int nf = 0; nf < 8; ++nf) {
            v16bf vb = cat8(*(const v8bf*)&VTs[(nf * 16 + ln) * VT_LD + g * 8],
                            *(const v8bf*)&VTs[(nf * 16 + ln) * VT_LD + 16 + g * 8]);
            o[nf] = wmma_bf16(pf, vb, o[nf]);
        }
        __syncthreads();
    }

    // Normalize and write context (bf16) for the output projection
    #pragma unroll
    for (int r = 0; r < 8; ++r) {
        int row = q0 + g * 8 + r;
        float inv = 1.0f / li[r];
        bf16* cp = ctx + ((size_t)(b * SEQ + row)) * EMBED + h * HDIM;
        #pragma unroll
        for (int nf = 0; nf < 8; ++nf)
            cp[nf * 16 + ln] = (bf16)(o[nf][r] * inv);
    }
}

// ---------------------------------------------------------------------------
// Host-side orchestration
// ---------------------------------------------------------------------------
extern "C" void kernel_launch(void* const* d_in, const int* in_sizes, int n_in,
                              void* d_out, int out_size, void* d_ws, size_t ws_size,
                              hipStream_t stream) {
    (void)in_sizes; (void)n_in; (void)out_size; (void)ws_size;

    const float* x    = (const float*)d_in[0];   // [B,S,E]
    const float* mask = (const float*)d_in[1];   // [B,S]
    const float* Wq   = (const float*)d_in[2];   // [E,E]
    const float* Wk   = (const float*)d_in[3];
    const float* Wv   = (const float*)d_in[4];
    const float* Wo   = (const float*)d_in[5];
    float* out = (float*)d_out;                  // [B,S,E] f32

    const long nX = (long)ROWS * EMBED;          // 8,388,608
    const long nW = (long)EMBED * EMBED;         // 4,194,304

    // Workspace layout (bf16 everywhere)
    char* w = (char*)d_ws;
    bf16* xb   = (bf16*)w;                     w += nX * 2;
    bf16* Wqb  = (bf16*)w;                     w += nW * 2;
    bf16* Wkb  = (bf16*)w;                     w += nW * 2;
    bf16* Wvb  = (bf16*)w;                     w += nW * 2;
    bf16* Wob  = (bf16*)w;                     w += nW * 2;
    bf16* Qb   = (bf16*)w;                     w += nX * 2;
    bf16* Kb   = (bf16*)w;                     w += nX * 2;
    bf16* Vb   = (bf16*)w;                     w += nX * 2;
    bf16* ctxb = (bf16*)w;

    // 1) Convert inputs to bf16
    cvt_f32_bf16<<<(int)(nX / 1024), 256, 0, stream>>>(x,  xb,  nX);
    cvt_f32_bf16<<<(int)(nW / 1024), 256, 0, stream>>>(Wq, Wqb, nW);
    cvt_f32_bf16<<<(int)(nW / 1024), 256, 0, stream>>>(Wk, Wkb, nW);
    cvt_f32_bf16<<<(int)(nW / 1024), 256, 0, stream>>>(Wv, Wvb, nW);
    cvt_f32_bf16<<<(int)(nW / 1024), 256, 0, stream>>>(Wo, Wob, nW);

    // 2) Q/K/V projections: [4096,2048] = x . W^T
    dim3 gg(EMBED / 128, ROWS / 128);
    gemm_nt<false><<<gg, 256, 0, stream>>>(xb, Wqb, Qb, ROWS, EMBED, EMBED);
    gemm_nt<false><<<gg, 256, 0, stream>>>(xb, Wkb, Kb, ROWS, EMBED, EMBED);
    gemm_nt<false><<<gg, 256, 0, stream>>>(xb, Wvb, Vb, ROWS, EMBED, EMBED);

    // 3) RoPE on Q,K: B*S*H*64 pair-threads
    long nRope = (long)BATCH * SEQ * HEADS * 64;
    rope_kernel<<<(int)(nRope / 256), 256, 0, stream>>>(Qb, Kb);

    // 4) Flash attention -> ctx (bf16)
    dim3 fg(SEQ / 128, BATCH * HEADS);
    flash_attn<<<fg, 256, 0, stream>>>(Qb, Kb, Vb, mask, ctxb);

    // 5) Output projection -> f32 d_out
    gemm_nt<true><<<gg, 256, 0, stream>>>(ctxb, Wob, out, ROWS, EMBED, EMBED);
}
